// LayerOutScalar_24378234372712
// MI455X (gfx1250) — matrix operand order repaired and checked by reading, not compiled, not run on hardware
//
#include <hip/hip_runtime.h>
#include <hip/hip_bf16.h>

// ---------------------------------------------------------------------------
// out = tanh(x @ W + b)   M = N = K = 4096, f32 in / f32 out
// bf16 WMMA path: f32 -> bf16 (RNE) packed in integer regs while staging
// tiles in LDS; accumulate f32 with v_wmma_f32_16x16x32_bf16.
// Block tile 256x128 (8 waves, each 64x64 = 4x4 WMMA accumulators).
// ---------------------------------------------------------------------------

typedef __attribute__((ext_vector_type(16))) __bf16 bf16x16;
typedef __attribute__((ext_vector_type(8)))  __bf16 bf16x8;
typedef __attribute__((ext_vector_type(8)))  float  f32x8;

static constexpr int NK     = 4096;
static constexpr int TILE_M = 256;
static constexpr int TILE_N = 128;
static constexpr int TILE_K = 32;
static constexpr int LDA    = TILE_K + 8;  // bf16 elems per LDS row (bank-conflict pad)

// round-to-nearest-even f32 -> bf16, result in high 16 bits of u32
__device__ __forceinline__ unsigned bfround(float f) {
    union { float f; unsigned u; } c; c.f = f;
    return c.u + 0x7FFFu + ((c.u >> 16) & 1u);
}
// pack two bf16 (lo -> bits[15:0], hi -> bits[31:16])
__device__ __forceinline__ unsigned pack2(float lo, float hi) {
    return (bfround(hi) & 0xFFFF0000u) | (bfround(lo) >> 16);
}

union FragU { bf16x16 v; bf16x8 h[2]; };

// A fragment (16x32 bf16, MxK): lane L holds row L%16.
// lanes 0-15:  VGPR0-3 = K0..7,  VGPR4-7 = K16..23
// lanes 16-31: VGPR0-3 = K8..15, VGPR4-7 = K24..31
__device__ __forceinline__ bf16x16 load_frag_a(const __bf16* tile_row0, int lane) {
    const int row = lane & 15;
    const int k0  = (lane & 16) ? 8 : 0;
    const __bf16* p = tile_row0 + row * LDA + k0;
    FragU f;
    f.h[0] = *(const bf16x8*)(p);
    f.h[1] = *(const bf16x8*)(p + 16);
    return f.v;
}

// B fragment (32x16 bf16, KxN), Bs stored [n][k]: lane L holds col L%16,
// lanes 0-15 K=0..7,16..23; lanes 16-31 K=8..15,24..31.
__device__ __forceinline__ bf16x16 load_frag_b(const __bf16* tile_col0, int lane) {
    const int col = lane & 15;
    const int k0  = (lane & 16) ? 8 : 0;
    const __bf16* p = tile_col0 + col * LDA + k0;
    FragU f;
    f.h[0] = *(const bf16x8*)(p);
    f.h[1] = *(const bf16x8*)(p + 16);
    return f.v;
}

// Stage x tile: 256 rows x 32 K f32 -> bf16 LDS. 2048 float4 / 256 threads.
__device__ __forceinline__ void load_x_tile(const float* __restrict__ x,
                                            int mBase, int kBase,
                                            __bf16 (*As)[LDA], int tid) {
#pragma unroll
    for (int i = 0; i < 8; ++i) {
        const int f   = tid + i * 256;
        const int row = f >> 3;          // 8 float4 per 32-wide row
        const int k   = (f & 7) * 4;
        const float4 v = *(const float4*)(x + (size_t)(mBase + row) * NK + kBase + k);
        uint2 p;
        p.x = pack2(v.x, v.y);
        p.y = pack2(v.z, v.w);
        *(uint2*)(&As[row][k]) = p;      // ds_store_b64
    }
}

// Stage W tile: 32 K-rows x 128 N f32 -> bf16 LDS transposed to [n][k].
// Each work item covers a (k, k+1) row pair x 4 columns -> 4 ds_store_b32.
__device__ __forceinline__ void load_w_tile(const float* __restrict__ W,
                                            int kBase, int nBase,
                                            __bf16 (*Bs)[LDA], int tid) {
#pragma unroll
    for (int i = 0; i < 2; ++i) {
        const int q = tid + i * 256;     // 512 work items total
        const int k = (q >> 5) * 2;      // k-pair
        const int n = (q & 31) * 4;
        const float4 v0 = *(const float4*)(W + (size_t)(kBase + k)     * NK + nBase + n);
        const float4 v1 = *(const float4*)(W + (size_t)(kBase + k + 1) * NK + nBase + n);
        *(unsigned*)(&Bs[n + 0][k]) = pack2(v0.x, v1.x);
        *(unsigned*)(&Bs[n + 1][k]) = pack2(v0.y, v1.y);
        *(unsigned*)(&Bs[n + 2][k]) = pack2(v0.z, v1.z);
        *(unsigned*)(&Bs[n + 3][k]) = pack2(v0.w, v1.w);
    }
}

// branch-free tanh: 1 - 2/(exp2(2*log2(e)*a) + 1); saturates correctly at +-inf
__device__ __forceinline__ float fast_tanh(float a) {
    const float t = __builtin_amdgcn_exp2f(a * 2.8853900817779268f);
    const float r = __builtin_amdgcn_rcpf(t + 1.0f);
    return __builtin_fmaf(-2.0f, r, 1.0f);
}

__global__ void __launch_bounds__(256)
gemm_bias_tanh_wmma(const float* __restrict__ x,
                    const float* __restrict__ W,
                    const float* __restrict__ bias,
                    float* __restrict__ out) {
    __shared__ __bf16 As[2][TILE_M][LDA];   // 40 KB
    __shared__ __bf16 Bs[2][TILE_N][LDA];   // 20 KB

    const int tid  = threadIdx.x;
    const int lane = tid & 31;
    const int wave = tid >> 5;
    const int wm   = wave & 3;   // 4 waves along M (64 each)
    const int wn   = wave >> 2;  // 2 waves along N (64 each)

    const int mBase = blockIdx.y * TILE_M;
    const int nBase = blockIdx.x * TILE_N;

    f32x8 c[4][4] = {};  // wave subtile: 64(M) x 64(N)

    load_x_tile(x, mBase, 0, As[0], tid);
    load_w_tile(W, 0, nBase, Bs[0], tid);
    __syncthreads();

    int buf = 0;
    constexpr int KT = NK / TILE_K;
    for (int kt = 0; kt < KT; ++kt) {
        // Prefetch k+2 tiles into cache (global_prefetch_b8).
        if (kt + 2 < KT) {
            const int kb2 = (kt + 2) * TILE_K;
            __builtin_prefetch(x + (size_t)(mBase + (tid >> 3)) * NK + kb2 + (tid & 7) * 4, 0, 1);
            __builtin_prefetch(W + (size_t)(kb2 + ((tid >> 5) * 2)) * NK + nBase + (tid & 31) * 4, 0, 1);
        }
        // Stage k+1 tiles into the other buffer (overlaps with WMMA below).
        if (kt + 1 < KT) {
            load_x_tile(x, mBase, (kt + 1) * TILE_K, As[buf ^ 1], tid);
            load_w_tile(W, (kt + 1) * TILE_K, nBase, Bs[buf ^ 1], tid);
        }

        bf16x16 a[4], b[4];
#pragma unroll
        for (int mi = 0; mi < 4; ++mi)
            a[mi] = load_frag_a(&As[buf][wm * 64 + mi * 16][0], lane);
#pragma unroll
        for (int ni = 0; ni < 4; ++ni)
            b[ni] = load_frag_b(&Bs[buf][wn * 64 + ni * 16][0], lane);

#pragma unroll
        for (int mi = 0; mi < 4; ++mi)
#pragma unroll
            for (int ni = 0; ni < 4; ++ni)
                c[mi][ni] = __builtin_amdgcn_wmma_f32_16x16x32_bf16(
                    false, a[mi], false, b[ni], (short)0, c[mi][ni], false, false);

        __syncthreads();
        buf ^= 1;
    }

    // Epilogue: bias + tanh, per 16x16 f32 C/D layout:
    //   VGPR v, lanes 0-15:  M=v,   N=lane
    //   VGPR v, lanes 16-31: M=v+8, N=lane-16
    const int ncol = lane & 15;
    const int rhi  = (lane >> 4) * 8;
#pragma unroll
    for (int ni = 0; ni < 4; ++ni) {
        const int col = nBase + wn * 64 + ni * 16 + ncol;
        const float bj = bias[col];
#pragma unroll
        for (int mi = 0; mi < 4; ++mi) {
            const int rbase = mBase + wm * 64 + mi * 16 + rhi;
            f32x8 acc = c[mi][ni];
#pragma unroll
            for (int v = 0; v < 8; ++v) {
                out[(size_t)(rbase + v) * NK + col] = fast_tanh(acc[v] + bj);
            }
        }
    }
}

extern "C" void kernel_launch(void* const* d_in, const int* in_sizes, int n_in,
                              void* d_out, int out_size, void* d_ws, size_t ws_size,
                              hipStream_t stream) {
    (void)in_sizes; (void)n_in; (void)out_size; (void)d_ws; (void)ws_size;
    const float* x  = (const float*)d_in[0];
    const float* W  = (const float*)d_in[1];
    const float* b  = (const float*)d_in[2];
    float* out      = (float*)d_out;

    dim3 grid(NK / TILE_N, NK / TILE_M);  // (32, 16)
    gemm_bias_tanh_wmma<<<grid, 256, 0, stream>>>(x, W, b, out);
}
